// QWenAttention_72885595013594
// MI455X (gfx1250) — matrix operand order repaired and checked by reading, not compiled, not run on hardware
//
#include <hip/hip_runtime.h>
#include <stdint.h>
#include <stddef.h>

// ---------------------------------------------------------------------------
// Types for WMMA fragments (gfx1250: v_wmma_f32_16x16x32_bf16)
// ---------------------------------------------------------------------------
typedef __attribute__((ext_vector_type(16))) __bf16 bf16x16;
typedef __attribute__((ext_vector_type(8)))  float  v8f;

struct alignas(16) B128 { unsigned int x[4]; };

union FragU {
  bf16x16        v;
  B128           u[2];
  unsigned short s[16];
};

__device__ __forceinline__ v8f v8f_zero() {
  v8f z;
#pragma unroll
  for (int i = 0; i < 8; ++i) z[i] = 0.0f;
  return z;
}

__device__ __forceinline__ v8f wmma_bf16(bf16x16 a, bf16x16 b, v8f c) {
  // D = A(16x32 bf16) * B(32x16 bf16) + C(16x16 f32)
  return __builtin_amdgcn_wmma_f32_16x16x32_bf16(
      /*neg_a=*/false, a, /*neg_b=*/false, b,
      /*c_mod=*/(short)0, c, /*reuse_a=*/false, /*reuse_b=*/false);
}

__device__ __forceinline__ unsigned short f2bf(float f) {
  unsigned u = __float_as_uint(f);
  unsigned r = u + 0x7FFFu + ((u >> 16) & 1u);   // round-to-nearest-even
  return (unsigned short)(r >> 16);
}
__device__ __forceinline__ float bf2f(unsigned short h) {
  return __uint_as_float(((unsigned)h) << 16);
}

// ---------------------------------------------------------------------------
// Problem constants (from the reference)
// ---------------------------------------------------------------------------
#define BATCH   2
#define SEQ     2048
#define HID     2048
#define NH      16
#define HD      128
#define H3      6144
#define ROWS    (BATCH * SEQ)     // 4096

// ---------------------------------------------------------------------------
// Kernel 1: fp32 -> bf16 conversion
// ---------------------------------------------------------------------------
__global__ __launch_bounds__(256) void cvt_f32_bf16(
    const float* __restrict__ in, unsigned short* __restrict__ out, int n) {
  int i = blockIdx.x * 256 + threadIdx.x;
  if (i < n) out[i] = f2bf(in[i]);
}

// ---------------------------------------------------------------------------
// Kernel 2: bf16 GEMM with WMMA. C[M,N] = A[M,K] x B[K,N] + bias
// Block tile 128x256 (256 threads = 8 waves), wave tile 64x64 (16 accums),
// K-step 32. LDS stride 40 elements (80B) keeps fragment reads 16B aligned.
// 16 WMMAs per 16 ds_load_b128 per K-step per wave.
// ---------------------------------------------------------------------------
#define LDA 40

template <bool OUT_BF16>
__global__ __launch_bounds__(256) void gemm_bf16_wmma(
    const unsigned short* __restrict__ A, const unsigned short* __restrict__ B,
    const float* __restrict__ bias, float* __restrict__ Cf,
    unsigned short* __restrict__ Cbf, int M, int N, int K) {
  __shared__ unsigned short As[128 * LDA];   // [m][k]
  __shared__ unsigned short Bs[256 * LDA];   // transposed: [n][k]

  const int tid   = threadIdx.x;
  const int lane  = tid & 31;
  const int wid   = tid >> 5;
  const int waveM = wid & 1;         // 0..1
  const int waveN = wid >> 1;        // 0..3
  const int m0b   = blockIdx.y * 128;
  const int n0b   = blockIdx.x * 256;
  const int l16   = lane & 15;
  const int lhalf = lane >> 4;
  const int aklo  = lhalf ? 8 : 0;   // A-frag: K = aklo+0..7 and aklo+16..23
  const int bklo  = lhalf ? 8 : 0;   // B-frag: K = bklo+0..7 and bklo+16..23

  v8f acc[4][4];
#pragma unroll
  for (int mt = 0; mt < 4; ++mt)
#pragma unroll
    for (int nt = 0; nt < 4; ++nt) acc[mt][nt] = v8f_zero();

  for (int k0 = 0; k0 < K; k0 += 32) {
    // A tile 128x32: 512 16-byte chunks, 2 per thread
#pragma unroll
    for (int c = 0; c < 2; ++c) {
      int chunk = tid + c * 256;
      int row = chunk >> 2;
      int cp  = chunk & 3;
      *(B128*)(&As[row * LDA + cp * 8]) =
          *(const B128*)(A + (unsigned)(m0b + row) * (unsigned)K + k0 + cp * 8);
    }
    // B tile 32x256, transposed into Bs[n][k]: 512 chunks of 16 n, 2/thread
#pragma unroll
    for (int c = 0; c < 2; ++c) {
      int chunk = tid + c * 256;
      int k = chunk >> 4;            // 0..31
      int n = (chunk & 15) * 16;     // 0..240
      FragU t;
      const B128* g = (const B128*)(B + (unsigned)(k0 + k) * (unsigned)N + n0b + n);
      t.u[0] = g[0];
      t.u[1] = g[1];
#pragma unroll
      for (int e = 0; e < 16; ++e) Bs[(n + e) * LDA + k] = t.s[e];
    }
    __syncthreads();

    bf16x16 aF[4];
#pragma unroll
    for (int mt = 0; mt < 4; ++mt) {
      int row = waveM * 64 + mt * 16 + l16;
      FragU f;
      f.u[0] = *(const B128*)(&As[row * LDA + aklo]);
      f.u[1] = *(const B128*)(&As[row * LDA + aklo + 16]);
      aF[mt] = f.v;
    }
    bf16x16 bF[4];
#pragma unroll
    for (int nt = 0; nt < 4; ++nt) {
      int nrow = waveN * 64 + nt * 16 + l16;
      FragU f;
      f.u[0] = *(const B128*)(&Bs[nrow * LDA + bklo]);
      f.u[1] = *(const B128*)(&Bs[nrow * LDA + bklo + 16]);
      bF[nt] = f.v;
    }
#pragma unroll
    for (int mt = 0; mt < 4; ++mt)
#pragma unroll
      for (int nt = 0; nt < 4; ++nt)
        acc[mt][nt] = wmma_bf16(aF[mt], bF[nt], acc[mt][nt]);
    __syncthreads();
  }

  // Epilogue: bias + store. D layout: row = i + lhalf*8, col = l16.
  // All offsets fit in 32 bits (max 4096*6144 < 2^31).
#pragma unroll
  for (int mt = 0; mt < 4; ++mt) {
#pragma unroll
    for (int nt = 0; nt < 4; ++nt) {
      unsigned col = (unsigned)(n0b + waveN * 64 + nt * 16 + l16);
      float bv = bias[col];
      unsigned row0 = (unsigned)(m0b + waveM * 64 + mt * 16 + lhalf * 8);
      unsigned ofs = row0 * (unsigned)N + col;
#pragma unroll
      for (int i = 0; i < 8; ++i) {
        float v = acc[mt][nt][i] + bv;
        if (OUT_BF16) Cbf[ofs] = f2bf(v);
        else          Cf[ofs]  = v;
        ofs += (unsigned)N;
      }
    }
  }
}

// ---------------------------------------------------------------------------
// Kernel 3: RoPE + reshape. qkv [B,S,3H] bf16 -> Q,K [B,h,S,hd] bf16,
// V transposed [B,h,hd,S] bf16 (so PV B-fragments are contiguous).
// logn scaling == 1 for S == 2048 (pos1 never exceeds SEQ_LENGTH).
// ---------------------------------------------------------------------------
__global__ __launch_bounds__(256) void rope_reshape(
    const unsigned short* __restrict__ qkv, const float* __restrict__ cosb,
    const float* __restrict__ sinb, unsigned short* __restrict__ Q,
    unsigned short* __restrict__ Kb, unsigned short* __restrict__ Vt) {
  int idx = blockIdx.x * 256 + threadIdx.x;          // [b, s, h, d]
  if (idx >= BATCH * SEQ * NH * HD) return;
  int d = idx & 127;
  int h = (idx >> 7) & 15;
  int s = (idx >> 11) & 2047;
  int b = idx >> 22;
  unsigned rowoff = (unsigned)(b * SEQ + s) * H3;
  int col = h * HD + d;
  float q = bf2f(qkv[rowoff + col]);
  float k = bf2f(qkv[rowoff + HID + col]);
  float v = bf2f(qkv[rowoff + 2 * HID + col]);
  int   dp  = (d < 64) ? d + 64 : d - 64;
  float sgn = (d < 64) ? -1.0f : 1.0f;
  float qp = bf2f(qkv[rowoff + h * HD + dp]);
  float kp = bf2f(qkv[rowoff + HID + h * HD + dp]);
  float c  = cosb[s * HD + d];
  float sn = sinb[s * HD + d];
  float qr = q * c + sgn * qp * sn;
  float kr = k * c + sgn * kp * sn;
  unsigned bh = (unsigned)(b * NH + h);
  Q[(bh * SEQ + s) * HD + d]  = f2bf(qr);
  Kb[(bh * SEQ + s) * HD + d] = f2bf(kr);
  Vt[(bh * HD + d) * SEQ + s] = f2bf(v);
}

// ---------------------------------------------------------------------------
// Kernel 4: causal flash attention. One wave (32 threads) per 16-row q tile.
// Per 32-key step: 8 WMMAs for scores, online softmax (shfl_xor row
// reductions across 16-lane halves), P staged through LDS to A-layout,
// 8 WMMAs for P*V. Output written bf16 in [B,S,nh*hd] for the final GEMM.
// ---------------------------------------------------------------------------
__global__ __launch_bounds__(32) void flash_attn_wmma(
    const unsigned short* __restrict__ Q, const unsigned short* __restrict__ Kb,
    const unsigned short* __restrict__ Vt, unsigned short* __restrict__ O) {
  __shared__ unsigned short plds[16 * LDA];

  const int lane  = threadIdx.x & 31;
  const int l16   = lane & 15;
  const int lhalf = lane >> 4;
  const int aklo  = lhalf ? 8 : 0;
  const int bklo  = lhalf ? 8 : 0;

  int bid = blockIdx.x;
  int qt  = bid & 127;            // q tile (16 rows)
  int h   = (bid >> 7) & 15;
  int b   = bid >> 11;
  unsigned bh = (unsigned)(b * NH + h);
  const unsigned short* Qbase = Q  + (bh * SEQ + (unsigned)qt * 16) * HD;
  const unsigned short* Kbase = Kb + bh * SEQ * HD;
  const unsigned short* Vbase = Vt + bh * HD * SEQ;

  // Q A-fragments over the 4 hd chunks of 32
  bf16x16 qF[4];
#pragma unroll
  for (int kc = 0; kc < 4; ++kc) {
    const unsigned short* p = Qbase + (unsigned)l16 * HD + kc * 32;
    FragU f;
    f.u[0] = *(const B128*)(p + aklo);
    f.u[1] = *(const B128*)(p + aklo + 16);
    qF[kc] = f.v;
  }

  v8f o[8];
#pragma unroll
  for (int t = 0; t < 8; ++t) o[t] = v8f_zero();
  float mrow[8], lrow[8];
#pragma unroll
  for (int i = 0; i < 8; ++i) { mrow[i] = -__builtin_inff(); lrow[i] = 0.0f; }

  const float scale = 0.08838834764831845f;   // 1/sqrt(128)
  const int nj = (qt >> 1) + 1;               // key blocks of 32 up to diagonal

  for (int j = 0; j < nj; ++j) {
    int kv0 = j * 32;
    v8f s0 = v8f_zero(), s1 = v8f_zero();
#pragma unroll
    for (int kc = 0; kc < 4; ++kc) {
      const unsigned short* p0 = Kbase + (unsigned)(kv0 + l16) * HD + kc * 32 + bklo;
      const unsigned short* p1 = Kbase + (unsigned)(kv0 + 16 + l16) * HD + kc * 32 + bklo;
      FragU f0, f1;
      f0.u[0] = *(const B128*)(p0); f0.u[1] = *(const B128*)(p0 + 16);
      f1.u[0] = *(const B128*)(p1); f1.u[1] = *(const B128*)(p1 + 16);
      s0 = wmma_bf16(qF[kc], f0.v, s0);
      s1 = wmma_bf16(qF[kc], f1.v, s1);
    }
    // online softmax: D layout row = qt*16 + i + lhalf*8, col = kv0 + l16 (+16)
    int col0 = kv0 + l16;
    int col1 = col0 + 16;
    float alpha[8];
#pragma unroll
    for (int i = 0; i < 8; ++i) {
      int row = qt * 16 + i + lhalf * 8;
      float a0 = (col0 <= row) ? s0[i] * scale : -__builtin_inff();
      float a1 = (col1 <= row) ? s1[i] * scale : -__builtin_inff();
      float t = fmaxf(a0, a1);
#pragma unroll
      for (int off = 1; off < 16; off <<= 1)
        t = fmaxf(t, __shfl_xor(t, off, 32));
      float mnew = fmaxf(mrow[i], t);
      alpha[i] = __expf(mrow[i] - mnew);
      float p0 = __expf(a0 - mnew);
      float p1 = __expf(a1 - mnew);
      s0[i] = p0; s1[i] = p1;
      float rs = p0 + p1;
#pragma unroll
      for (int off = 1; off < 16; off <<= 1)
        rs += __shfl_xor(rs, off, 32);
      lrow[i] = lrow[i] * alpha[i] + rs;
      mrow[i] = mnew;
    }
#pragma unroll
    for (int t = 0; t < 8; ++t)
#pragma unroll
      for (int i = 0; i < 8; ++i) o[t][i] *= alpha[i];

    // P: D-layout -> LDS (row-major 16x32) -> A-layout fragment
#pragma unroll
    for (int i = 0; i < 8; ++i) {
      int r = i + lhalf * 8;
      plds[r * LDA + l16]      = f2bf(s0[i]);
      plds[r * LDA + 16 + l16] = f2bf(s1[i]);
    }
    __syncthreads();
    FragU pf;
    {
      const unsigned short* p = plds + l16 * LDA;
      pf.u[0] = *(const B128*)(p + aklo);
      pf.u[1] = *(const B128*)(p + aklo + 16);
    }
    __syncthreads();

    // O += P(16x32) * V(32xhd): 8 hd column tiles
#pragma unroll
    for (int t = 0; t < 8; ++t) {
      const unsigned short* p = Vbase + (unsigned)(t * 16 + l16) * SEQ + kv0 + bklo;
      FragU vf;
      vf.u[0] = *(const B128*)(p);
      vf.u[1] = *(const B128*)(p + 16);
      o[t] = wmma_bf16(pf.v, vf.v, o[t]);
    }
  }

  // normalize (multiply by per-row reciprocal) and store [B,S,nh*hd] bf16
  float rinv[8];
#pragma unroll
  for (int i = 0; i < 8; ++i) rinv[i] = __frcp_rn(lrow[i]);
#pragma unroll
  for (int t = 0; t < 8; ++t) {
#pragma unroll
    for (int i = 0; i < 8; ++i) {
      unsigned row = (unsigned)(qt * 16 + i + lhalf * 8);
      float v = o[t][i] * rinv[i];
      O[((unsigned)b * SEQ + row) * HID + h * HD + t * 16 + l16] = f2bf(v);
    }
  }
}

// ---------------------------------------------------------------------------
// Workspace layout (bytes)
// ---------------------------------------------------------------------------
#define OFF_X    ((size_t)0)            // hidden bf16     4096*2048*2 = 16 MiB
#define OFF_WA   ((size_t)16777216)     // W_attn bf16     2048*6144*2 = 24 MiB
#define OFF_WP   ((size_t)41943040)     // W_proj bf16     2048*2048*2 =  8 MiB
#define OFF_QKV  ((size_t)50331648)     // qkv bf16        4096*6144*2 = 48 MiB
#define OFF_Q    ((size_t)100663296)    // Q bf16                        16 MiB
#define OFF_K    ((size_t)117440512)    // K bf16                        16 MiB
#define OFF_VT   ((size_t)134217728)    // V^T bf16                      16 MiB
#define OFF_ATT  ((size_t)150994944)    // attn out bf16                 16 MiB

extern "C" void kernel_launch(void* const* d_in, const int* in_sizes, int n_in,
                              void* d_out, int out_size, void* d_ws, size_t ws_size,
                              hipStream_t stream) {
  const float* hidden = (const float*)d_in[0];
  const float* cosb   = (const float*)d_in[1];
  const float* sinb   = (const float*)d_in[2];
  // d_in[3] = causal mask (computed analytically; unused)
  const float* W_attn = (const float*)d_in[4];
  const float* b_attn = (const float*)d_in[5];
  const float* W_proj = (const float*)d_in[6];
  const float* b_proj = (const float*)d_in[7];
  float* out = (float*)d_out;

  char* ws = (char*)d_ws;
  unsigned short* Xbf  = (unsigned short*)(ws + OFF_X);
  unsigned short* WAbf = (unsigned short*)(ws + OFF_WA);
  unsigned short* WPbf = (unsigned short*)(ws + OFF_WP);
  unsigned short* QKV  = (unsigned short*)(ws + OFF_QKV);
  unsigned short* Qb   = (unsigned short*)(ws + OFF_Q);
  unsigned short* Kbuf = (unsigned short*)(ws + OFF_K);
  unsigned short* Vt   = (unsigned short*)(ws + OFF_VT);
  unsigned short* Att  = (unsigned short*)(ws + OFF_ATT);

  // 1) fp32 -> bf16 conversions
  {
    int n;
    n = ROWS * HID;   cvt_f32_bf16<<<(n + 255) / 256, 256, 0, stream>>>(hidden, Xbf, n);
    n = HID * H3;     cvt_f32_bf16<<<(n + 255) / 256, 256, 0, stream>>>(W_attn, WAbf, n);
    n = HID * HID;    cvt_f32_bf16<<<(n + 255) / 256, 256, 0, stream>>>(W_proj, WPbf, n);
  }

  // 2) QKV projection: [4096,2048] x [2048,6144] + b_attn -> bf16
  {
    dim3 grid(H3 / 256, ROWS / 128);
    gemm_bf16_wmma<true><<<grid, 256, 0, stream>>>(Xbf, WAbf, b_attn, nullptr, QKV,
                                                   ROWS, H3, HID);
  }

  // 3) RoPE + reshape (logn scale == 1 at S == 2048)
  {
    int n = BATCH * SEQ * NH * HD;
    rope_reshape<<<(n + 255) / 256, 256, 0, stream>>>(QKV, cosb, sinb, Qb, Kbuf, Vt);
  }

  // 4) causal flash attention: one wave per (b, h, 16-row q tile)
  {
    int nblk = BATCH * NH * (SEQ / 16);   // 4096
    flash_attn_wmma<<<nblk, 32, 0, stream>>>(Qb, Kbuf, Vt, Att);
  }

  // 5) output projection: [4096,2048] x [2048,2048] + b_proj -> f32 d_out
  {
    dim3 grid(HID / 256, ROWS / 128);
    gemm_bf16_wmma<false><<<grid, 256, 0, stream>>>(Att, WPbf, b_proj, out, nullptr,
                                                    ROWS, HID, HID);
  }
}
